// STC_LIF_45621142618290
// MI455X (gfx1250) — matrix operand order repaired and checked by reading, not compiled, not run on hardware
//
#include <hip/hip_runtime.h>

// ---------------------------------------------------------------------------
// STC-LIF recurrence on MI455X (gfx1250, wave32).
//
// 512 dependent steps -> minimize per-step critical path:
//  * 16-row stripe per block, 4 waves x 32 columns; W^T register-resident
//    (64 VGPRs/wave) -> 8 back-to-back v_wmma_f32_16x16x32_f16, two
//    interleaved accumulator chains, only A-side LDS loads.
//  * Elementwise LIF update done DIRECTLY in the WMMA C-fragment layout:
//    no gamma LDS round trip; v_tanh_f32 feeds straight off accumulators
//    (bias is one scalar/lane/tile since C-frag N == lane%16).
//  * Spikes exchanged via double-buffered LDS with a SPLIT barrier:
//    s_wait_dscnt 0 + s_barrier_signal -1 right after spike stores;
//    s_barrier_wait -1 right before next step's A loads. x loads, prefetch
//    and out stores live outside the synchronized region.
// ---------------------------------------------------------------------------

typedef __attribute__((ext_vector_type(16))) _Float16 v16h;
typedef __attribute__((ext_vector_type(8)))  _Float16 v8h;   // 16B -> ds_load_b128
typedef __attribute__((ext_vector_type(8)))  float    v8f;

#define NSTEP 512          // STEPS * T
#define BSZ   1024
#define DIM   128
#define SS_P  136          // halfs: padded pitch for spike rows (bank spread)

__device__ __forceinline__ float fast_tanh(float v) {
#if __has_builtin(__builtin_amdgcn_tanhf)
    return __builtin_amdgcn_tanhf(v);          // v_tanh_f32
#elif __has_builtin(__builtin_amdgcn_tanh_f32)
    return __builtin_amdgcn_tanh_f32(v);
#else
    return tanhf(v);
#endif
}

#if __has_builtin(__builtin_amdgcn_s_barrier_signal) && \
    __has_builtin(__builtin_amdgcn_s_barrier_wait)
#define LDS_FENCE()  asm volatile("s_wait_dscnt 0x0" ::: "memory")
#define BAR_SIGNAL() __builtin_amdgcn_s_barrier_signal(-1)
#define BAR_WAIT()   __builtin_amdgcn_s_barrier_wait(-1)
#else
#define LDS_FENCE()  ((void)0)
#define BAR_SIGNAL() __syncthreads()
#define BAR_WAIT()   ((void)0)
#endif

__global__ __launch_bounds__(128) void stc_lif_kernel(
    const float* __restrict__ x,     // [512,1024,128] (flat-identical reshape)
    const float* __restrict__ mem0,  // [1024,128]
    const float* __restrict__ sp0,   // [1024,128]
    const float* __restrict__ Wgs,   // [128,128]
    const float* __restrict__ bgs,   // [128]
    float*       __restrict__ out)   // [512,1024,128]
{
    // Only LDS: double-buffered stripe spikes (f16), 8704 B.
    __shared__ __align__(16) _Float16 SS[2][16][SS_P];

    const int tid   = threadIdx.x;
    const int wv    = tid >> 5;               // wave 0..3
    const int l     = tid & 31;
    const int rbase = blockIdx.x * 16;        // stripe rows
    const int cbase = wv * 32;                // this wave's 32 columns

    // ---- B fragments, register-resident forever.  B[k][n] = W[n_g][k] ----
    v16h b[2][4];
#pragma unroll
    for (int j2 = 0; j2 < 2; ++j2)
#pragma unroll
        for (int kc = 0; kc < 4; ++kc) {
            v16h f;
#pragma unroll
            for (int n = 0; n < 16; ++n)
                f[n] = (_Float16)Wgs[(cbase + j2 * 16 + (l & 15)) * DIM + kc * 32 + ((l >> 4) << 3) + (n & 7) + ((n >> 3) << 4)];
            b[j2][kc] = f;
        }

    // ---- C-fragment coordinates (ISA 7.12.2): reg r -> M=crow+r, N=ccol ----
    const int crow = (l >> 4) << 3;
    const int ccol = l & 15;
    const int M    = l & 15;                  // A-frag row for this lane
    const int K0   = (l >> 4) << 3;           // A-frag K split (0-7/16-23 vs 8-15/24-31)

    // per-lane bias: one scalar per N-tile (C-frag column is fixed per lane)
    const float bias0 = bgs[cbase + ccol];
    const float bias1 = bgs[cbase + 16 + ccol];

    // membrane state + initial spikes, all in C-fragment layout
    float m[2][8];
#pragma unroll
    for (int j2 = 0; j2 < 2; ++j2) {
        const int col = cbase + j2 * 16 + ccol;
#pragma unroll
        for (int r = 0; r < 8; ++r) {
            const int row = crow + r;
            m[j2][r] = mem0[(size_t)(rbase + row) * DIM + col];
            SS[0][row][col] = (_Float16)sp0[(size_t)(rbase + row) * DIM + col];
        }
    }
    LDS_FENCE();
    BAR_SIGNAL();                              // pairs with step 0's BAR_WAIT

    for (int st = 0; st < NSTEP; ++st) {
        const int p = st & 1;
        const size_t base = (size_t)st * (BSZ * DIM);

        // x loads are GEMM-independent: issue before waiting on the barrier.
        float xv[2][8];
#pragma unroll
        for (int j2 = 0; j2 < 2; ++j2)
#pragma unroll
            for (int r = 0; r < 8; ++r)
                xv[j2][r] = __builtin_nontemporal_load(
                    x + base + (size_t)(rbase + crow + r) * DIM +
                    cbase + j2 * 16 + ccol);

        if (st + 1 < NSTEP) {
            // one prefetch: lane-varying address covers the full 16x512B stripe
            __builtin_prefetch(x + base + (size_t)(BSZ * DIM) +
                                   (size_t)(rbase + (l & 15)) * DIM +
                                   cbase + ((l >> 4) << 4), 0, 3);
        }

        BAR_WAIT();                            // all waves published SS[p]

        // ---- A fragments from the stripe's spike buffer (parity p) ----
        v16h a[4];
#pragma unroll
        for (int kc = 0; kc < 4; ++kc) {
            union { v16h v; v8h h[2]; } u;
            u.h[0] = *(const v8h*)&SS[p][M][kc * 32 + K0];
            u.h[1] = *(const v8h*)&SS[p][M][kc * 32 + K0 + 16];
            a[kc] = u.v;
        }

        // ---- GEMM: 2 N-tiles x 4 K-chunks, B in registers ----
        v8f acc[2];
#pragma unroll
        for (int j2 = 0; j2 < 2; ++j2) {
            acc[j2] = (v8f){};
#pragma unroll
            for (int kc = 0; kc < 4; ++kc)
                acc[j2] = __builtin_amdgcn_wmma_f32_16x16x32_f16(
                              false, a[kc], false, b[j2][kc], (short)0,
                              acc[j2], false, false);
        }

        // ---- LIF update directly on accumulators (C-fragment layout) ----
        float sv[2][8];
#pragma unroll
        for (int j2 = 0; j2 < 2; ++j2) {
            const float bj  = (j2 == 0) ? bias0 : bias1;
            const int   col = cbase + j2 * 16 + ccol;
#pragma unroll
            for (int r = 0; r < 8; ++r) {
                const float gam = 0.5f * (1.0f + fast_tanh(acc[j2][r] + bj));
                float mm = m[j2][r] + xv[j2][r] * gam;
                const float sp = (mm >= 1.0f) ? 1.0f : 0.0f;   // zif(m-1)*THRESH
                m[j2][r] = mm * (1.0f - sp);                   // soft reset
                sv[j2][r] = sp;
                SS[p ^ 1][crow + r][col] = (_Float16)sp;       // next step's A
            }
        }

        LDS_FENCE();
        if (st + 1 < NSTEP) BAR_SIGNAL();      // release partner waves ASAP

        // out stores are outside the synchronized region (fire-and-forget)
#pragma unroll
        for (int j2 = 0; j2 < 2; ++j2)
#pragma unroll
            for (int r = 0; r < 8; ++r)
                __builtin_nontemporal_store(
                    sv[j2][r],
                    out + base + (size_t)(rbase + crow + r) * DIM +
                        cbase + j2 * 16 + ccol);
    }
}

extern "C" void kernel_launch(void* const* d_in, const int* in_sizes, int n_in,
                              void* d_out, int out_size, void* d_ws, size_t ws_size,
                              hipStream_t stream) {
    const float* x    = (const float*)d_in[0];
    const float* mem0 = (const float*)d_in[1];
    const float* sp0  = (const float*)d_in[2];
    const float* Wgs  = (const float*)d_in[3];
    const float* bgs  = (const float*)d_in[4];
    // d_in[5]/d_in[6] (W_gt_w / W_gt_b) are dead compute in this branch.
    float* out = (float*)d_out;

    // 64 blocks x 128 threads: one 16-row stripe per block, 4 waves x 32 cols.
    stc_lif_kernel<<<dim3(64), dim3(128), 0, stream>>>(x, mem0, sp0, Wgs, bgs, out);
}
